// CrossAttention_55774445306047
// MI455X (gfx1250) — matrix operand order repaired and checked by reading, not compiled, not run on hardware
//
#include <hip/hip_runtime.h>

#define BB   2
#define NQ   4096
#define NKV  4096
#define CDIM 256
#define NH   8
#define HD   32
#define QSCALE 0.17677669529663687f   // 32^-0.5, folded into Q at projection time

typedef __attribute__((ext_vector_type(16))) _Float16 v16h;
typedef __attribute__((ext_vector_type(8)))  _Float16 v8h;
typedef __attribute__((ext_vector_type(8)))  float    v8f;
typedef __attribute__((ext_vector_type(4)))  float    v4f;

__device__ inline v8f wmma16(v16h a, v16h b, v8f c) {
  // D = A(16x32 f16) * B(32x16 f16) + C(16x16 f32)
  return __builtin_amdgcn_wmma_f32_16x16x32_f16(false, a, false, b, (short)0, c,
                                                false, false);
}

// ---- CDNA5 async global->LDS copy (16B per lane), tracked by ASYNCcnt ------
__device__ inline void async_copy_b128(unsigned lds_byte_off, const void* gsrc) {
  unsigned long long ga = (unsigned long long)(uintptr_t)gsrc;
  asm volatile("global_load_async_to_lds_b128 %0, %1, off"
               :
               : "v"(lds_byte_off), "v"(ga)
               : "memory");
}
__device__ inline void wait_async0() {
  asm volatile("s_wait_asynccnt 0x0" ::: "memory");
}

// ---- A fragment (16 rows x K=32) built from an f32 row-major source --------
// A-frag layout: lane&15 = M row; elems = K {ksel..ksel+15},
// ksel = 16*(lane>=16).
__device__ inline v16h a_frag_f32(const float* __restrict__ src, int ld,
                                  int row, int kb, int lane) {
  int m = lane & 15;
  int ksel = (lane >> 4) << 4;
  const float* p = src + (size_t)(row + m) * ld + kb + ksel;
  v4f x0 = *(const v4f*)(p);
  v4f x1 = *(const v4f*)(p + 4);
  v4f x2 = *(const v4f*)(p + 8);
  v4f x3 = *(const v4f*)(p + 12);
  v16h a;
#pragma unroll
  for (int j = 0; j < 4; ++j) {
    a[j]      = (_Float16)x0[j];
    a[4 + j]  = (_Float16)x1[j];
    a[8 + j]  = (_Float16)x2[j];
    a[12 + j] = (_Float16)x3[j];
  }
  return a;
}

// ---- A fragment from an f16 row-major source (row stride ld halves) --------
__device__ inline v16h a_frag_f16(const _Float16* __restrict__ src, int ld,
                                  int row, int kb, int lane) {
  int m = lane & 15;
  int ksel = (lane >> 4) << 4;
  const _Float16* p = src + (size_t)(row + m) * ld + kb + ksel;
  v8h lo = *(const v8h*)(p);
  v8h hi = *(const v8h*)(p + 8);
  return __builtin_shufflevector(lo, hi, 0, 1, 2, 3, 4, 5, 6, 7,
                                 8, 9, 10, 11, 12, 13, 14, 15);
}

// ---- B fragment from k-contiguous "[n][k]" f16 storage ---------------------
// B-frag layout: lane&15 = N col; lanes<16 hold K 0..15, lanes>=16 K 16..31,
// element j = K koff+j  -> one contiguous 32B load.
__device__ inline v16h b_frag(const _Float16* __restrict__ srcT, int ldk,
                              int n, int kb, int lane) {
  int nn = n + (lane & 15);
  int koff = (lane >> 4) << 4;
  return *(const v16h*)(srcT + (size_t)nn * ldk + kb + koff);
}

// ============================================================================
// Kernel 1: transpose + f32->f16 convert all weights into [n][k] layout.
// ============================================================================
__global__ __launch_bounds__(256) void k_cvt_w(const float* __restrict__ w_q,
                                               const float* __restrict__ w_kv,
                                               const float* __restrict__ w_out,
                                               _Float16* __restrict__ wqT,
                                               _Float16* __restrict__ wkvT,
                                               _Float16* __restrict__ woT) {
  int idx = blockIdx.x * 256 + threadIdx.x;            // 0 .. 262143
  if (idx < 65536) {                                   // w_q: 256x256
    int n = idx >> 8, k = idx & 255;
    wqT[idx] = (_Float16)w_q[k * 256 + n];
  } else if (idx < 65536 + 131072) {                   // w_kv: 256x512
    int j = idx - 65536;
    int n = j >> 8, k = j & 255;                       // n in [0,512)
    wkvT[j] = (_Float16)w_kv[k * 512 + n];
  } else {                                             // w_out: 256x256
    int j = idx - 196608;
    int n = j >> 8, k = j & 255;
    woT[j] = (_Float16)w_out[k * 256 + n];
  }
}

// ============================================================================
// Kernel 2: Q projection. [8192,256] x [256,256] -> Qh f16 [b,h,n,d], *scale.
// Wave tile = 16 x 64. Block = 8 waves = 128 rows. grid = (64, 4).
// ============================================================================
__global__ __launch_bounds__(256) void k_proj_q(const float* __restrict__ query,
                                                const _Float16* __restrict__ wqT,
                                                _Float16* __restrict__ Qh) {
  int lane = threadIdx.x & 31, wave = threadIdx.x >> 5;
  int rowbase = blockIdx.x * 128 + wave * 16;
  int colbase = blockIdx.y * 64;
  v8f acc[4];
#pragma unroll
  for (int t = 0; t < 4; ++t)
#pragma unroll
    for (int j = 0; j < 8; ++j) acc[t][j] = 0.0f;

  for (int kb = 0; kb < CDIM; kb += 32) {
    v16h a = a_frag_f32(query, CDIM, rowbase, kb, lane);
#pragma unroll
    for (int t = 0; t < 4; ++t)
      acc[t] = wmma16(a, b_frag(wqT, CDIM, colbase + t * 16, kb, lane), acc[t]);
  }
  int hsel = (lane >> 4) << 3;
#pragma unroll
  for (int t = 0; t < 4; ++t)
#pragma unroll
    for (int r = 0; r < 8; ++r) {
      int m = rowbase + r + hsel;                 // global row in [0,8192)
      int c = colbase + t * 16 + (lane & 15);     // out channel
      int b = m >> 12, n = m & 4095;
      int hh = c >> 5, d = c & 31;
      Qh[(((size_t)b * NH + hh) * NQ + n) * HD + d] =
          (_Float16)(acc[t][r] * QSCALE);
    }
}

// ============================================================================
// Kernel 3: KV projection. [8192,256] x [256,512].
// K -> Kh f16 [b,h,n,d] (row-major, A-frag friendly in d)
// V -> Vt f16 [b,h,d,n] (transposed, B-frag friendly in kv)
// grid = (64, 8).
// ============================================================================
__global__ __launch_bounds__(256) void k_proj_kv(const float* __restrict__ kvsrc,
                                                 const _Float16* __restrict__ wkvT,
                                                 _Float16* __restrict__ Kh,
                                                 _Float16* __restrict__ Vt) {
  int lane = threadIdx.x & 31, wave = threadIdx.x >> 5;
  int rowbase = blockIdx.x * 128 + wave * 16;
  int colbase = blockIdx.y * 64;
  v8f acc[4];
#pragma unroll
  for (int t = 0; t < 4; ++t)
#pragma unroll
    for (int j = 0; j < 8; ++j) acc[t][j] = 0.0f;

  for (int kb = 0; kb < CDIM; kb += 32) {
    v16h a = a_frag_f32(kvsrc, CDIM, rowbase, kb, lane);
#pragma unroll
    for (int t = 0; t < 4; ++t)
      acc[t] = wmma16(a, b_frag(wkvT, CDIM, colbase + t * 16, kb, lane), acc[t]);
  }
  int hsel = (lane >> 4) << 3;
#pragma unroll
  for (int t = 0; t < 4; ++t)
#pragma unroll
    for (int r = 0; r < 8; ++r) {
      int m = rowbase + r + hsel;
      int c = colbase + t * 16 + (lane & 15);     // in [0,512)
      int b = m >> 12, n = m & 4095;
      int kvsel = c >> 8, hh = (c >> 5) & 7, d = c & 31;
      _Float16 v = (_Float16)acc[t][r];
      if (kvsel == 0)
        Kh[(((size_t)b * NH + hh) * NKV + n) * HD + d] = v;
      else
        Vt[(((size_t)b * NH + hh) * HD + d) * NKV + n] = v;
    }
}

// ============================================================================
// Kernel 4: fused flash attention.
// Transposed-score trick: S^T = K_chunk(16x32) @ Q^T(32x16) so the exp'd
// scores land directly in A-fragment layout for P@V — no lane transpose.
// Block = 8 waves, all on one (b,h); K/V 32-kv chunks staged into LDS with
// CDNA5 async copies (global_load_async_to_lds_b128 / ASYNCcnt), double-
// buffered so the next chunk's DMA overlaps the current chunk's WMMAs.
// Each wave owns 64 q rows (4 accumulator pairs). grid = B*H*(NQ/512) = 128.
// ============================================================================
__global__ __launch_bounds__(256) void k_attn(const _Float16* __restrict__ Qh,
                                              const _Float16* __restrict__ Kh,
                                              const _Float16* __restrict__ Vt,
                                              _Float16* __restrict__ AttnO) {
  __shared__ __align__(64) _Float16 Ks[2][32 * 32];   // [buf][kv][d]
  __shared__ __align__(64) _Float16 Vs[2][32 * 32];   // [buf][d][kv]

  const int t = threadIdx.x;
  const int lane = t & 31, wave = t >> 5;
  const int bh = blockIdx.x >> 3, qblk = blockIdx.x & 7;
  const int b = bh >> 3, hh = bh & 7;
  const int qwb = qblk * 512 + wave * 64;
  const int hsel = (lane >> 4) << 3;
  const int koff16 = (lane >> 4) << 4;

  const _Float16* Kbase = Kh + (size_t)bh * NKV * HD;
  const _Float16* Vbase = Vt + (size_t)bh * HD * NKV;

  // One async b128 per thread per chunk: waves 0-3 stage K (2KB contiguous),
  // waves 4-7 stage V^T (32 rows x 64B).
  auto issue_chunk = [&](int kv, int buf) {
    if (t < 128) {
      async_copy_b128((unsigned)(uintptr_t)(&Ks[buf][t * 8]),
                      Kbase + (size_t)kv * HD + t * 8);
    } else {
      int tt = t - 128;
      int d = tt >> 2, part = tt & 3;
      async_copy_b128((unsigned)(uintptr_t)(&Vs[buf][d * 32 + part * 8]),
                      Vbase + (size_t)d * NKV + kv + part * 8);
    }
  };

  // Q^T B-fragments for the 4 q tiles (held in registers whole kernel)
  v16h bq[4];
#pragma unroll
  for (int qt = 0; qt < 4; ++qt)
    bq[qt] = *(const v16h*)(Qh + ((size_t)bh * NQ + qwb + qt * 16 + (lane & 15)) * HD +
                            koff16);

  v8f o0[4], o1[4];
  float mprev[4], lsum[4];
#pragma unroll
  for (int qt = 0; qt < 4; ++qt) {
#pragma unroll
    for (int j = 0; j < 8; ++j) { o0[qt][j] = 0.0f; o1[qt][j] = 0.0f; }
    mprev[qt] = -3.0e38f;
    lsum[qt] = 0.0f;
  }

  issue_chunk(0, 0);   // prologue DMA

  for (int kv = 0; kv < NKV; kv += 32) {
    const int buf = (kv >> 5) & 1;
    wait_async0();        // this wave's DMA (into `buf`) has landed in LDS
    __syncthreads();      // all waves' DMAs landed; all reads of buf^1 done
    if (kv + 32 < NKV) issue_chunk(kv + 32, buf ^ 1);   // overlap next DMA

    v16h aK0 = a_frag_f16(Ks[buf], 32, 0, 0, lane);    // kv rows 0..15
    v16h aK1 = a_frag_f16(Ks[buf], 32, 16, 0, lane);   // kv rows 16..31
    const _Float16* pv0 = Vs[buf] + (lane & 15) * 32 + hsel;
    const _Float16* pv1 = Vs[buf] + ((lane & 15) + 16) * 32 + hsel;
    v16h bv0 = __builtin_shufflevector(*(const v8h*)pv0, *(const v8h*)(pv0 + 16), 0, 1, 2, 3, 4, 5, 6, 7, 8, 9, 10, 11, 12, 13, 14, 15);        // d 0..15
    v16h bv1 = __builtin_shufflevector(*(const v8h*)pv1, *(const v8h*)(pv1 + 16), 0, 1, 2, 3, 4, 5, 6, 7, 8, 9, 10, 11, 12, 13, 14, 15); // d 16..31

#pragma unroll
    for (int qt = 0; qt < 4; ++qt) {
      v8f z;
#pragma unroll
      for (int j = 0; j < 8; ++j) z[j] = 0.0f;
      v8f st0 = wmma16(aK0, bq[qt], z);   // S^T kv 0..15  x q 0..15
      v8f st1 = wmma16(aK1, bq[qt], z);   // S^T kv 16..31 x q 0..15

      // online softmax stats for q = lane&15 (replicated across halves)
      float mx = -3.0e38f;
#pragma unroll
      for (int j = 0; j < 8; ++j) mx = fmaxf(mx, fmaxf(st0[j], st1[j]));
      mx = fmaxf(mx, __shfl_xor(mx, 16, 32));
      float mnew = fmaxf(mprev[qt], mx);
      float alpha = __expf(mprev[qt] - mnew);

      v16h ap;           // P in A-fragment layout — no cross-lane movement
      float rs = 0.0f;
#pragma unroll
      for (int j = 0; j < 8; ++j) {
        float p0 = __expf(st0[j] - mnew);
        float p1 = __expf(st1[j] - mnew);
        ap[j] = (_Float16)p0;
        ap[8 + j] = (_Float16)p1;
        rs += p0 + p1;
      }
      rs += __shfl_xor(rs, 16, 32);
      lsum[qt] = lsum[qt] * alpha + rs;
      mprev[qt] = mnew;

      // rescale O rows (row index lives in VGPR elems -> broadcast via shfl)
#pragma unroll
      for (int r = 0; r < 8; ++r) {
        float ar = __shfl(alpha, r + hsel, 32);
        o0[qt][r] *= ar;
        o1[qt][r] *= ar;
      }
      o0[qt] = wmma16(ap, bv0, o0[qt]);
      o1[qt] = wmma16(ap, bv1, o1[qt]);
    }
  }

  // ---- normalize and write O as f16 [b, q, h*32+d] for the out-projection --
#pragma unroll
  for (int qt = 0; qt < 4; ++qt) {
    float linv = 1.0f / lsum[qt];
#pragma unroll
    for (int r = 0; r < 8; ++r) {
      float le = __shfl(linv, r + hsel, 32);
      int q = qwb + qt * 16 + r + hsel;
      int d = lane & 15;
      size_t base = ((size_t)b * NQ + q) * CDIM + hh * HD;
      AttnO[base + d] = (_Float16)(o0[qt][r] * le);
      AttnO[base + d + 16] = (_Float16)(o1[qt][r] * le);
    }
  }
}

// ============================================================================
// Kernel 5: out projection + bias. [8192,256]f16 x [256,256] -> f32 d_out.
// grid = (64, 4).
// ============================================================================
__global__ __launch_bounds__(256) void k_proj_out(const _Float16* __restrict__ AttnO,
                                                  const _Float16* __restrict__ woT,
                                                  const float* __restrict__ b_out,
                                                  float* __restrict__ out) {
  int lane = threadIdx.x & 31, wave = threadIdx.x >> 5;
  int rowbase = blockIdx.x * 128 + wave * 16;
  int colbase = blockIdx.y * 64;
  v8f acc[4];
#pragma unroll
  for (int t = 0; t < 4; ++t)
#pragma unroll
    for (int j = 0; j < 8; ++j) acc[t][j] = 0.0f;

  for (int kb = 0; kb < CDIM; kb += 32) {
    v16h a = a_frag_f16(AttnO, CDIM, rowbase, kb, lane);
#pragma unroll
    for (int t = 0; t < 4; ++t)
      acc[t] = wmma16(a, b_frag(woT, CDIM, colbase + t * 16, kb, lane), acc[t]);
  }
  int hsel = (lane >> 4) << 3;
#pragma unroll
  for (int t = 0; t < 4; ++t)
#pragma unroll
    for (int r = 0; r < 8; ++r) {
      int m = rowbase + r + hsel;
      int c = colbase + t * 16 + (lane & 15);
      out[(size_t)m * CDIM + c] = acc[t][r] + b_out[c];
    }
}

// ============================================================================
extern "C" void kernel_launch(void* const* d_in, const int* in_sizes, int n_in,
                              void* d_out, int out_size, void* d_ws, size_t ws_size,
                              hipStream_t stream) {
  const float* query = (const float*)d_in[0];   // [2,4096,256]
  const float* keyv  = (const float*)d_in[1];   // [2,4096,256]
  const float* w_q   = (const float*)d_in[2];   // [256,256]
  const float* w_kv  = (const float*)d_in[3];   // [256,512]
  const float* w_out = (const float*)d_in[4];   // [256,256]
  const float* b_out = (const float*)d_in[5];   // [256]
  float* out = (float*)d_out;                   // [2,4096,256] f32

  // ---- workspace carve-up (all offsets 64B-aligned) ----
  char* ws = (char*)d_ws;
  _Float16* wqT   = (_Float16*)(ws);                       // 256*256*2   = 128 KiB
  _Float16* wkvT  = (_Float16*)(ws + 131072);              // 512*256*2   = 256 KiB
  _Float16* woT   = (_Float16*)(ws + 393216);              // 256*256*2   = 128 KiB
  _Float16* Qh    = (_Float16*)(ws + 524288);              // 2*8*4096*32 = 4 MiB
  _Float16* Kh    = (_Float16*)(ws + 4718592);             // 4 MiB
  _Float16* Vt    = (_Float16*)(ws + 8912896);             // 4 MiB
  _Float16* AttnO = (_Float16*)(ws + 13107200);            // 4 MiB  (tot ~16.5 MiB)

  k_cvt_w<<<1024, 256, 0, stream>>>(w_q, w_kv, w_out, wqT, wkvT, woT);
  k_proj_q<<<dim3(64, 4), 256, 0, stream>>>(query, wqT, Qh);
  k_proj_kv<<<dim3(64, 8), 256, 0, stream>>>(keyv, wkvT, Kh, Vt);
  k_attn<<<128, 256, 0, stream>>>(Qh, Kh, Vt, AttnO);
  k_proj_out<<<dim3(64, 4), 256, 0, stream>>>(AttnO, woT, b_out, out);
}
